// FeatureEnhance_66125316489511
// MI455X (gfx1250) — matrix operand order repaired and checked by reading, not compiled, not run on hardware
//
#include <hip/hip_runtime.h>

// ---------------------------------------------------------------------------
// CDNA5 (gfx1250) FeatureEnhance: all contractions via v_wmma_f32_16x16x32_bf16
// (bf16 operands, f32 accumulate). Intermediates padded to multiples of 32 so
// interior tiles stage via float4 global loads + b128 LDS stores, and WMMA
// fragments are read as two b128 LDS loads per operand (ISA 7.12.2 layout).
// ---------------------------------------------------------------------------

typedef __bf16 v16bf __attribute__((ext_vector_type(16)));
typedef float  v8f   __attribute__((ext_vector_type(8)));

#define KSTEP 32
#define LROW  40          // LDS row stride in bf16 elems (80B, 16B-aligned)

union FragU  { uint4 q[2]; v16bf v; };
union PackU  { __bf16 h[16]; uint4 q[2]; };

// Generic batched GEMM:
//   D[b,m,n] = alpha * (sum_k A(b,m,k) * X(b,k,n)) + bias[m] + S[b,m,n]
// TA: A element (m,k) at k*ldA+m ; TB: X element (k,n) at n*ldB+k.
// alpha read from *alpha_ptr when non-null (device scalar gamma).
template <int TA, int TB>
__global__ __launch_bounds__(32)
void wmma_gemm_kernel(const float* __restrict__ A, long ldA, long strideA,
                      const float* __restrict__ Bm, long ldB, long strideB,
                      const float* __restrict__ bias,
                      const float* __restrict__ S, long ldS, long strideS,
                      float* __restrict__ D, long ldD, long strideD,
                      float alpha_const, const float* __restrict__ alpha_ptr,
                      int M, int N, int K)
{
    __shared__ __bf16 As[16][LROW];     // A tile, [m][k]
    __shared__ __bf16 Bs[16][LROW];     // B tile, K-major: [n][k]

    const int lane  = threadIdx.x;      // 0..31 (wave32)
    const int tn    = blockIdx.x * 16;
    const int tm    = blockIdx.y * 16;
    const int b     = blockIdx.z;
    const int idx16 = lane & 15;
    const int hf    = lane >> 4;

    const float* Ab = A  + (size_t)b * strideA;
    const float* Xb = Bm + (size_t)b * strideB;

    const bool fullMN = (tm + 16 <= M) && (tn + 16 <= N);

    v8f acc = {};

    for (int k0 = 0; k0 < K; k0 += KSTEP) {
        if (fullMN && (k0 + KSTEP <= K)) {
            // ======== fast path: vectorized staging, branch-free ========
            if (TA == 0) {
                // rows contiguous along k: lane -> (row=idx16, k half=hf)
                const float* src = Ab + (size_t)(tm + idx16) * ldA + k0 + hf * 16;
                float4 f0 = *(const float4*)(src + 0);
                float4 f1 = *(const float4*)(src + 4);
                float4 f2 = *(const float4*)(src + 8);
                float4 f3 = *(const float4*)(src + 12);
                PackU pk;
                pk.h[0]=(__bf16)f0.x; pk.h[1]=(__bf16)f0.y; pk.h[2]=(__bf16)f0.z; pk.h[3]=(__bf16)f0.w;
                pk.h[4]=(__bf16)f1.x; pk.h[5]=(__bf16)f1.y; pk.h[6]=(__bf16)f1.z; pk.h[7]=(__bf16)f1.w;
                pk.h[8]=(__bf16)f2.x; pk.h[9]=(__bf16)f2.y; pk.h[10]=(__bf16)f2.z; pk.h[11]=(__bf16)f2.w;
                pk.h[12]=(__bf16)f3.x; pk.h[13]=(__bf16)f3.y; pk.h[14]=(__bf16)f3.z; pk.h[15]=(__bf16)f3.w;
                uint4* dst = (uint4*)&As[idx16][hf * 16];
                dst[0] = pk.q[0]; dst[1] = pk.q[1];
            } else {
                // columns contiguous along m: lane -> k column
                const float* src = Ab + (size_t)(k0 + lane) * ldA + tm;
                float4 f0 = *(const float4*)(src + 0);
                float4 f1 = *(const float4*)(src + 4);
                float4 f2 = *(const float4*)(src + 8);
                float4 f3 = *(const float4*)(src + 12);
                As[0][lane]=(__bf16)f0.x;  As[1][lane]=(__bf16)f0.y;
                As[2][lane]=(__bf16)f0.z;  As[3][lane]=(__bf16)f0.w;
                As[4][lane]=(__bf16)f1.x;  As[5][lane]=(__bf16)f1.y;
                As[6][lane]=(__bf16)f1.z;  As[7][lane]=(__bf16)f1.w;
                As[8][lane]=(__bf16)f2.x;  As[9][lane]=(__bf16)f2.y;
                As[10][lane]=(__bf16)f2.z; As[11][lane]=(__bf16)f2.w;
                As[12][lane]=(__bf16)f3.x; As[13][lane]=(__bf16)f3.y;
                As[14][lane]=(__bf16)f3.z; As[15][lane]=(__bf16)f3.w;
            }
            if (TB == 0) {
                // rows contiguous along n: lane -> k row; scatter into [n][k]
                const float* src = Xb + (size_t)(k0 + lane) * ldB + tn;
                float4 f0 = *(const float4*)(src + 0);
                float4 f1 = *(const float4*)(src + 4);
                float4 f2 = *(const float4*)(src + 8);
                float4 f3 = *(const float4*)(src + 12);
                Bs[0][lane]=(__bf16)f0.x;  Bs[1][lane]=(__bf16)f0.y;
                Bs[2][lane]=(__bf16)f0.z;  Bs[3][lane]=(__bf16)f0.w;
                Bs[4][lane]=(__bf16)f1.x;  Bs[5][lane]=(__bf16)f1.y;
                Bs[6][lane]=(__bf16)f1.z;  Bs[7][lane]=(__bf16)f1.w;
                Bs[8][lane]=(__bf16)f2.x;  Bs[9][lane]=(__bf16)f2.y;
                Bs[10][lane]=(__bf16)f2.z; Bs[11][lane]=(__bf16)f2.w;
                Bs[12][lane]=(__bf16)f3.x; Bs[13][lane]=(__bf16)f3.y;
                Bs[14][lane]=(__bf16)f3.z; Bs[15][lane]=(__bf16)f3.w;
            } else {
                // columns contiguous along k: lane -> (n=idx16, k half=hf)
                const float* src = Xb + (size_t)(tn + idx16) * ldB + k0 + hf * 16;
                float4 f0 = *(const float4*)(src + 0);
                float4 f1 = *(const float4*)(src + 4);
                float4 f2 = *(const float4*)(src + 8);
                float4 f3 = *(const float4*)(src + 12);
                PackU pk;
                pk.h[0]=(__bf16)f0.x; pk.h[1]=(__bf16)f0.y; pk.h[2]=(__bf16)f0.z; pk.h[3]=(__bf16)f0.w;
                pk.h[4]=(__bf16)f1.x; pk.h[5]=(__bf16)f1.y; pk.h[6]=(__bf16)f1.z; pk.h[7]=(__bf16)f1.w;
                pk.h[8]=(__bf16)f2.x; pk.h[9]=(__bf16)f2.y; pk.h[10]=(__bf16)f2.z; pk.h[11]=(__bf16)f2.w;
                pk.h[12]=(__bf16)f3.x; pk.h[13]=(__bf16)f3.y; pk.h[14]=(__bf16)f3.z; pk.h[15]=(__bf16)f3.w;
                uint4* dst = (uint4*)&Bs[idx16][hf * 16];
                dst[0] = pk.q[0]; dst[1] = pk.q[1];
            }
        } else {
            // ======== edge path: guarded scalar staging ========
            for (int t = lane; t < 16 * KSTEP; t += 32) {
                int row = t >> 5, kk = t & 31;
                int m = tm + row, k = k0 + kk;
                float v = 0.f;
                if (m < M && k < K)
                    v = Ab[TA ? ((size_t)k * ldA + m) : ((size_t)m * ldA + k)];
                As[row][kk] = (__bf16)v;
            }
            for (int t = lane; t < KSTEP * 16; t += 32) {
                int kk = t >> 4, nn = t & 15;
                int k = k0 + kk, n = tn + nn;
                float v = 0.f;
                if (k < K && n < N)
                    v = Xb[TB ? ((size_t)n * ldB + k) : ((size_t)k * ldB + nn)];
                Bs[nn][kk] = (__bf16)v;
            }
        }
        __syncthreads();

        // ---- fragments: two b128 LDS loads per operand (ISA 7.12.2) ----
        // A: elems 0..7 <- K=8*hf..8*hf+7 ; elems 8..15 <- K=16+8*hf..+7
        FragU fa;
        fa.q[0] = *(const uint4*)&As[idx16][8 * hf];
        fa.q[1] = *(const uint4*)&As[idx16][16 + 8 * hf];
        // B: elems 0..7 <- K=8*hf..8*hf+7 ; elems 8..15 <- K=16+8*hf..+7 (K-major in LDS)
        FragU fb;
        fb.q[0] = *(const uint4*)&Bs[idx16][8 * hf];
        fb.q[1] = *(const uint4*)&Bs[idx16][16 + 8 * hf];

        acc = __builtin_amdgcn_wmma_f32_16x16x32_bf16(
                  false, fa.v, false, fb.v, (short)0, acc, false, false);
        __syncthreads();
    }

    const float alpha = alpha_ptr ? *alpha_ptr : alpha_const;
    const int ncol = tn + idx16;
    if (ncol < N) {
        #pragma unroll
        for (int r = 0; r < 8; ++r) {            // C/D: VGPR r -> M = r + 8*hf
            int m = tm + r + 8 * hf;
            if (m < M) {
                float v = alpha * acc[r];
                if (bias) v += bias[m];
                if (S)    v += S[(size_t)b * strideS + (size_t)m * ldS + ncol];
                D[(size_t)b * strideD + (size_t)m * ldD + ncol] = v;
            }
        }
    }
}

// Copy [Bn,C,n] into padded [Bn,C,npad] with zeroed tail.
__global__ void pad_copy_kernel(float* __restrict__ dst, const float* __restrict__ src,
                                int n, int npad, long rows)
{
    long i = (long)blockIdx.x * blockDim.x + threadIdx.x;
    long total = rows * npad;
    if (i < total) {
        int p = (int)(i % npad);
        long r = i / npad;
        dst[i] = (p < n) ? src[r * n + p] : 0.f;
    }
}

// Row softmax over first L entries of each npad-strided row; zero pad tail.
__global__ void softmax_rows_kernel(float* __restrict__ a, int L, int npad, int rowsPerB)
{
    __shared__ float red[256];
    const int b = blockIdx.x / rowsPerB;
    const int r = blockIdx.x % rowsPerB;
    float* row = a + (size_t)b * npad * npad + (size_t)r * npad;
    const int tid = threadIdx.x;

    float m = -1e30f;
    for (int i = tid; i < L; i += blockDim.x) m = fmaxf(m, row[i]);
    red[tid] = m; __syncthreads();
    for (int s = blockDim.x >> 1; s > 0; s >>= 1) {
        if (tid < s) red[tid] = fmaxf(red[tid], red[tid + s]);
        __syncthreads();
    }
    m = red[0]; __syncthreads();

    float sum = 0.f;
    for (int i = tid; i < L; i += blockDim.x) {
        float e = __expf(row[i] - m);
        row[i] = e; sum += e;
    }
    red[tid] = sum; __syncthreads();
    for (int s = blockDim.x >> 1; s > 0; s >>= 1) {
        if (tid < s) red[tid] += red[tid + s];
        __syncthreads();
    }
    const float inv = 1.f / red[0];
    for (int i = tid; i < L; i += blockDim.x) row[i] *= inv;
    for (int i = L + tid; i < npad; i += blockDim.x) row[i] = 0.f;
}

// Channel-attention softmax: softmax(rowmax - e) == exp(rowmin - e)/sum.
__global__ void cam_softmax_kernel(float* __restrict__ a, int L)
{
    __shared__ float red[256];
    float* row = a + (size_t)blockIdx.x * L;
    const int tid = threadIdx.x;

    float mn = 1e30f;
    for (int i = tid; i < L; i += blockDim.x) mn = fminf(mn, row[i]);
    red[tid] = mn; __syncthreads();
    for (int s = blockDim.x >> 1; s > 0; s >>= 1) {
        if (tid < s) red[tid] = fminf(red[tid], red[tid + s]);
        __syncthreads();
    }
    mn = red[0]; __syncthreads();

    float sum = 0.f;
    for (int i = tid; i < L; i += blockDim.x) {
        float e = __expf(mn - row[i]);
        row[i] = e; sum += e;
    }
    red[tid] = sum; __syncthreads();
    for (int s = blockDim.x >> 1; s > 0; s >>= 1) {
        if (tid < s) red[tid] += red[tid + s];
        __syncthreads();
    }
    const float inv = 1.f / red[0];
    for (int i = tid; i < L; i += blockDim.x) row[i] *= inv;
}

// d[i] += a * x[i]
__global__ void axpy_kernel(float* __restrict__ d, const float* __restrict__ x,
                            float a, long N)
{
    long i = (long)blockIdx.x * blockDim.x + threadIdx.x;
    if (i < N) d[i] += a * x[i];
}

// 3x3 same-pad conv producing 18 offset channels; padded channel strides.
__global__ void offconv_kernel(const float* __restrict__ x, const float* __restrict__ w,
                               const float* __restrict__ bias, float* __restrict__ out,
                               int Bn, int H, int W, int npad)
{
    const int n = H * W;
    const long total = (long)Bn * 18 * n;
    for (long i = (long)blockIdx.x * blockDim.x + threadIdx.x; i < total;
         i += (long)gridDim.x * blockDim.x) {
        int p = (int)(i % n);
        int o = (int)((i / n) % 18);
        int b = (int)(i / ((long)18 * n));
        int y = p / W, xx = p % W;
        float acc = bias[o];
        const float* xb = x + (size_t)b * 256 * npad;
        const float* wo = w + (size_t)o * 256 * 9;
        for (int c = 0; c < 256; ++c) {
            const float* xc = xb + (size_t)c * npad;
            const float* wc = wo + c * 9;
            #pragma unroll
            for (int t = 0; t < 9; ++t) {
                int yy  = y  + t / 3 - 1;
                int xx2 = xx + t % 3 - 1;
                if (yy >= 0 && yy < H && xx2 >= 0 && xx2 < W)
                    acc += wc[t] * xc[yy * W + xx2];
            }
        }
        out[((size_t)b * 18 + o) * npad + p] = acc;
    }
}

// Deformable bilinear im2col: col[b][c*9+k][p] (padded strides).
__global__ void bilinear_col_kernel(const float* __restrict__ img,
                                    const float* __restrict__ off,
                                    float* __restrict__ col,
                                    int Bn, int H, int W, int npad)
{
    const int n = H * W;
    const long total = (long)Bn * 9 * n;
    for (long i = (long)blockIdx.x * blockDim.x + threadIdx.x; i < total;
         i += (long)gridDim.x * blockDim.x) {
        int p = (int)(i % n);
        int k = (int)((i / n) % 9);
        int b = (int)(i / ((long)9 * n));
        int y = p / W, x = p % W;

        float dy = off[((size_t)b * 18 + 2 * k)     * npad + p];
        float dx = off[((size_t)b * 18 + 2 * k + 1) * npad + p];
        float ys = (float)y + (float)(k / 3 - 1) + dy;
        float xs = (float)x + (float)(k % 3 - 1) + dx;

        float y0f = floorf(ys), x0f = floorf(xs);
        int   y0 = (int)y0f,    x0 = (int)x0f;
        int   y1 = y0 + 1,      x1 = x0 + 1;
        float wy = ys - y0f,    wx = xs - x0f;

        bool vy0 = (y0 >= 0 && y0 < H), vy1 = (y1 >= 0 && y1 < H);
        bool vx0 = (x0 >= 0 && x0 < W), vx1 = (x1 >= 0 && x1 < W);
        float w00 = (vy0 && vx0) ? (1.f - wy) * (1.f - wx) : 0.f;
        float w01 = (vy0 && vx1) ? (1.f - wy) * wx         : 0.f;
        float w10 = (vy1 && vx0) ? wy * (1.f - wx)         : 0.f;
        float w11 = (vy1 && vx1) ? wy * wx                 : 0.f;

        int yc0 = min(max(y0, 0), H - 1), yc1 = min(max(y1, 0), H - 1);
        int xc0 = min(max(x0, 0), W - 1), xc1 = min(max(x1, 0), W - 1);
        int i00 = yc0 * W + xc0, i01 = yc0 * W + xc1;
        int i10 = yc1 * W + xc0, i11 = yc1 * W + xc1;

        const float* imb = img + (size_t)b * 256 * npad;
        float* colp = col + (size_t)b * 2304 * npad + (size_t)k * npad + p;
        const long cstride = (long)9 * npad;
        for (int c = 0; c < 256; ++c) {
            const float* ic = imb + (size_t)c * npad;
            float v = w00 * ic[i00] + w01 * ic[i01] + w10 * ic[i10] + w11 * ic[i11];
            colp[(size_t)c * cstride] = v;
        }
    }
}

// ---------------------------------------------------------------------------
extern "C" void kernel_launch(void* const* d_in, const int* in_sizes, int n_in,
                              void* d_out, int out_size, void* d_ws, size_t ws_size,
                              hipStream_t stream)
{
    const int Bn = 8, C = 256, CQ = 32;
    const int Hs[2] = {15, 31};
    const int Ns[2] = {225, 961};
    const int NPs[2] = {256, 992};     // padded spatial dims (multiples of 32)
    const int npmax = 992;

    const float* feats[2][3];          // [f][scale], f=0: z, f=1: x
    for (int s = 0; s < 3; ++s) { feats[0][s] = (const float*)d_in[s];
                                  feats[1][s] = (const float*)d_in[3 + s]; }

    // params flattened per-scale in JAX pytree (sorted dict key) order:
    // bk, bq, bv, dw, gamma_c, gamma_p, off_b, off_w, wk, wq, wv
    auto P = [&](int s, int j) -> const float* {
        return (const float*)d_in[6 + s * 11 + j];
    };
    const int I_BK = 0, I_BQ = 1, I_BV = 2, I_DW = 3, I_GC = 4, I_GP = 5,
              I_OFFB = 6, I_OFFW = 7, I_WK = 8, I_WQ = 9, I_WV = 10;

    // --- workspace carve (floats), padded layouts ---
    float* ws = (float*)d_ws;
    size_t off = 0;
    auto take = [&](size_t nf) { float* p = ws + off; off += nf; return p; };
    float* fpZ  = take((size_t)Bn * C  * npmax);   // padded z feature
    float* fpX  = take((size_t)Bn * C  * npmax);   // padded x feature
    float* qb   = take((size_t)Bn * CQ * npmax);
    float* kb   = take((size_t)Bn * CQ * npmax);
    float* vb   = take((size_t)Bn * C  * npmax);
    float* attn = take((size_t)Bn * npmax * npmax);
    float* pamZ = take((size_t)Bn * C * npmax);
    float* pamX = take((size_t)Bn * C * npmax);
    float* camZ = take((size_t)Bn * C * C);
    float* camX = take((size_t)Bn * C * C);
    float* cmbZ = take((size_t)Bn * C * npmax);
    float* cmbX = take((size_t)Bn * C * npmax);
    float* offb = take((size_t)Bn * 18 * npmax);
    float* colb = take((size_t)Bn * (C * 9) * npmax);
    (void)ws_size; (void)in_sizes; (void)n_in; (void)out_size;

    auto gemm = [&](int tA, int tB,
                    const float* A, long ldA, long sA,
                    const float* X, long ldB, long sB,
                    const float* bias,
                    const float* S, long ldS, long sS,
                    float* D, long ldD, long sD,
                    float alpha, const float* aptr,
                    int M, int N, int K) {
        dim3 g((N + 15) / 16, (M + 15) / 16, Bn);
        if (tA == 0 && tB == 0)
            wmma_gemm_kernel<0,0><<<g, 32, 0, stream>>>(A, ldA, sA, X, ldB, sB, bias,
                                                        S, ldS, sS, D, ldD, sD,
                                                        alpha, aptr, M, N, K);
        else if (tA == 0 && tB == 1)
            wmma_gemm_kernel<0,1><<<g, 32, 0, stream>>>(A, ldA, sA, X, ldB, sB, bias,
                                                        S, ldS, sS, D, ldD, sD,
                                                        alpha, aptr, M, N, K);
        else
            wmma_gemm_kernel<1,0><<<g, 32, 0, stream>>>(A, ldA, sA, X, ldB, sB, bias,
                                                        S, ldS, sS, D, ldD, sD,
                                                        alpha, aptr, M, N, K);
    };

    const size_t zOutSz = (size_t)Bn * C * 225;   // 460800
    const size_t xOutSz = (size_t)Bn * C * 961;   // 1968128

    for (int s = 0; s < 3; ++s) {
        const float* gp = P(s, I_GP);
        const float* gc = P(s, I_GC);

        // --- padded feature copies + channel attention matrices ---
        for (int f = 0; f < 2; ++f) {
            const int n = Ns[f], np = NPs[f];
            float* fp  = (f == 0) ? fpZ  : fpX;
            float* cam = (f == 0) ? camZ : camX;
            long rows = (long)Bn * C;
            long tot = rows * np;
            pad_copy_kernel<<<(int)((tot + 255) / 256), 256, 0, stream>>>(
                fp, feats[f][s], n, np, rows);
            // e[b,c,d] = sum_p fp[c,p]*fp[d,p]  (zero pads => K = np ok)
            gemm(0, 1, fp, np, (long)C * np, fp, np, (long)C * np,
                 nullptr, nullptr, 0, 0,
                 cam, C, (long)C * C, 1.f, nullptr, C, C, np);
            cam_softmax_kernel<<<Bn * C, 256, 0, stream>>>(cam, C);
        }

        for (int f = 0; f < 2; ++f) {
            const int n = Ns[f], np = NPs[f], H = Hs[f], W = Hs[f];
            const float* fp = (f == 0) ? fpZ : fpX;
            float* pam  = (f == 0) ? pamZ : pamX;
            float* comb = (f == 0) ? cmbZ : cmbX;
            const float* camOwn = (f == 0) ? camZ : camX;
            const float* camOth = (f == 0) ? camX : camZ;

            // projections q,k (32ch), v (256ch); pad cols get bias (harmless)
            gemm(0, 0, P(s, I_WQ), C, 0, fp, np, (long)C * np, P(s, I_BQ),
                 nullptr, 0, 0, qb, np, (long)CQ * np, 1.f, nullptr, CQ, np, C);
            gemm(0, 0, P(s, I_WK), C, 0, fp, np, (long)C * np, P(s, I_BK),
                 nullptr, 0, 0, kb, np, (long)CQ * np, 1.f, nullptr, CQ, np, C);
            gemm(0, 0, P(s, I_WV), C, 0, fp, np, (long)C * np, P(s, I_BV),
                 nullptr, 0, 0, vb, np, (long)C * np, 1.f, nullptr, C, np, C);

            // attn[b,i,j] = sum_c q[c,i]*k[c,j]; softmax rows (+zero pad tail)
            gemm(1, 0, qb, np, (long)CQ * np, kb, np, (long)CQ * np,
                 nullptr, nullptr, 0, 0,
                 attn, np, (long)np * np, 1.f, nullptr, n, n, CQ);
            softmax_rows_kernel<<<Bn * n, 256, 0, stream>>>(attn, n, np, n);

            // pam = gamma_p * (v @ attn^T) + fp   (K = np: attn pads zeroed)
            gemm(0, 1, vb, np, (long)C * np, attn, np, (long)np * np,
                 nullptr, fp, np, (long)C * np,
                 pam, np, (long)C * np, 1.f, gp, C, np, np);

            // comb = gamma_c*(camOwn @ fp) + pam ; comb += gamma_c*(camOth @ fp)
            gemm(0, 0, camOwn, C, (long)C * C, fp, np, (long)C * np,
                 nullptr, pam, np, (long)C * np,
                 comb, np, (long)C * np, 1.f, gc, C, np, C);
            gemm(0, 0, camOth, C, (long)C * C, fp, np, (long)C * np,
                 nullptr, comb, np, (long)C * np,
                 comb, np, (long)C * np, 1.f, gc, C, np, C);
            {   // the two _cam_use residuals each add +feat
                long tot = (long)Bn * C * np;
                axpy_kernel<<<(int)((tot + 255) / 256), 256, 0, stream>>>(
                    comb, fp, 2.f, tot);
            }

            // --- deformable conv on comb ---
            {
                long tot = (long)Bn * 18 * n;
                offconv_kernel<<<(int)((tot + 255) / 256), 256, 0, stream>>>(
                    comb, P(s, I_OFFW), P(s, I_OFFB), offb, Bn, H, W, np);
            }
            {
                long tot = (long)Bn * 9 * n;
                bilinear_col_kernel<<<(int)((tot + 255) / 256), 256, 0, stream>>>(
                    comb, offb, colb, Bn, H, W, np);
            }
            float* outp = (float*)d_out +
                          ((f == 0) ? (size_t)s * zOutSz
                                    : 3 * zOutSz + (size_t)s * xOutSz);
            // out[b,o,p] = dw[o, c*9+k] @ col[b, c*9+k, p]  (output unpadded)
            gemm(0, 0, P(s, I_DW), C * 9, 0, colb, np, (long)(C * 9) * np,
                 nullptr, nullptr, 0, 0,
                 outp, n, (long)C * n, 1.f, nullptr, C, n, C * 9);
        }
    }
}
